// SparseConvBlock_57466662420876
// MI455X (gfx1250) — matrix operand order repaired and checked by reading, not compiled, not run on hardware
//
#include <hip/hip_runtime.h>
#include <hip/hip_bf16.h>

typedef __attribute__((ext_vector_type(16))) _Float16 v16h;
typedef __attribute__((ext_vector_type(8)))  _Float16 v8h;
typedef __attribute__((ext_vector_type(8)))  float    v8f;

#define GDIM   130      // (G + 2) as in the reference
#define CIN    32
#define COUT   64
#define KVOL   27
#define BN_EPS 1e-5f
#define CONV_WAVES 8
#define RED_BLOCKS 256

// ---------------- prep kernels ----------------

__global__ void init_grid_kernel(int* __restrict__ grid, int n) {
  int i = blockIdx.x * blockDim.x + threadIdx.x;
  for (; i < n; i += gridDim.x * blockDim.x) grid[i] = -1;
}

__global__ void scatter_grid_kernel(int* __restrict__ grid,
                                    const int* __restrict__ pos, int nIn) {
  int i = blockIdx.x * blockDim.x + threadIdx.x;
  if (i >= nIn) return;
  int x = pos[3 * i + 0], y = pos[3 * i + 1], z = pos[3 * i + 2];
  grid[(x * GDIM + y) * GDIM + z] = i;
}

// features -> f16, plus one trailing all-zero row for empty neighbors (idx==-1)
__global__ void cvt_features_kernel(const float* __restrict__ f,
                                    _Float16* __restrict__ f16, int nIn) {
  int tot = (nIn + 1) * CIN;
  int i = blockIdx.x * blockDim.x + threadIdx.x;
  for (; i < tot; i += gridDim.x * blockDim.x)
    f16[i] = (i < nIn * CIN) ? (_Float16)f[i] : (_Float16)0.0f;
}

// Repack weight[k][kk][n] (f32) into per-lane WMMA B fragments (f16):
// wB[((k*4 + t)*32 + lane)*16 + i] = w[k][ i + 16*(lane>>4) ][ t*16 + (lane&15) ]
// in contiguous K order, matching the A gather: lanes 0-15 hold K=0..15, lanes 16-31 K=16..31.
__global__ void repack_weights_kernel(const float* __restrict__ w,
                                      _Float16* __restrict__ wB) {
  int idx = blockIdx.x * blockDim.x + threadIdx.x;
  if (idx >= KVOL * 4 * 32 * 16) return;
  int i    = idx & 15;
  int lane = (idx >> 4) & 31;
  int t    = (idx >> 9) & 3;
  int k    = idx >> 11;
  int kk   = i + ((lane >> 4) << 4);
  int n    = t * 16 + (lane & 15);
  wB[idx] = (_Float16)w[(k * CIN + kk) * COUT + n];
}

// ---------------- main sparse-conv WMMA kernel ----------------

__global__ __launch_bounds__(CONV_WAVES * 32)
void sparse_conv_wmma_kernel(const int* __restrict__ grid,
                             const _Float16* __restrict__ feat16,
                             const _Float16* __restrict__ wB,
                             const int* __restrict__ out_pos,
                             float* __restrict__ out,
                             int nIn, int nOut, int ntiles) {
  // Stage all repacked B fragments (27 taps * 4 N-tiles * 32 lanes * 16 halves
  // = 110592 B) into LDS once per block. 320 KB WGP LDS -> 2 blocks/WGP.
  extern __shared__ _Float16 sB[];
  {
    const int nvec = KVOL * 4 * 32 * 16 / 8;   // v8h (16B) chunks
    const v8h* src = (const v8h*)wB;
    v8h* dst = (v8h*)sB;
    for (int i = threadIdx.x; i < nvec; i += blockDim.x) dst[i] = src[i];
  }
  __syncthreads();

  const int wave = threadIdx.x >> 5;
  const int lane = threadIdx.x & 31;
  const int tile = blockIdx.x * CONV_WAVES + wave;
  if (tile >= ntiles) return;

  const int rowBase = tile * 16;
  const int half = lane >> 4;       // which 16-lane half of the wave
  const int m    = lane & 15;       // A-matrix row / C-matrix column index

  // Clamp tail rows for the gather; tail stores are predicated below, so the
  // duplicated gather never pollutes the output (EXEC stays all-1 for WMMA).
  int row = rowBase + m;
  if (row > nOut - 1) row = nOut - 1;
  const int ox = out_pos[3 * row + 0];
  const int oy = out_pos[3 * row + 1];
  const int oz = out_pos[3 * row + 2];

  v8f acc0 = {}, acc1 = {}, acc2 = {}, acc3 = {};

#pragma unroll
  for (int dx = 0; dx < 3; ++dx)
#pragma unroll
    for (int dy = 0; dy < 3; ++dy)
#pragma unroll
      for (int dz = 0; dz < 3; ++dz) {
        const int k = (dx * 3 + dy) * 3 + dz;
        int nidx = grid[((ox + dx) * GDIM + (oy + dy)) * GDIM + (oz + dz)];
        int fi = (nidx < 0) ? nIn : nidx;   // -1 -> zero row

        // A fragment, contiguous K order (matches B repack):
        // lane<16 : halves 0-15 = K 0-15   (chunks 0,1)
        // lane>=16: halves 0-15 = K 16-31  (chunks 2,3)
        const v8h* fr = (const v8h*)(feat16 + (size_t)fi * CIN);
        v8h lo = fr[2 * half];
        v8h hi = fr[2 * half + 1];
        v16h a = __builtin_shufflevector(lo, hi, 0, 1, 2, 3, 4, 5, 6, 7,
                                                 8, 9, 10, 11, 12, 13, 14, 15);

        const v16h* bbase = ((const v16h*)sB) + (k * 4) * 32 + lane;
        acc0 = __builtin_amdgcn_wmma_f32_16x16x32_f16(false, a, false, bbase[0],
                                                      (short)0, acc0, false, false);
        acc1 = __builtin_amdgcn_wmma_f32_16x16x32_f16(false, a, false, bbase[32],
                                                      (short)0, acc1, false, false);
        acc2 = __builtin_amdgcn_wmma_f32_16x16x32_f16(false, a, false, bbase[64],
                                                      (short)0, acc2, false, false);
        acc3 = __builtin_amdgcn_wmma_f32_16x16x32_f16(false, a, false, bbase[96],
                                                      (short)0, acc3, false, false);
      }

  // C/D layout: VGPR r -> (M = r + 8*half, N = t*16 + m)
#pragma unroll
  for (int r = 0; r < 8; ++r) {
    int orow = rowBase + r + half * 8;
    if (orow < nOut) {
      float* o = out + (size_t)orow * COUT;
      o[0 * 16 + m] = acc0[r];
      o[1 * 16 + m] = acc1[r];
      o[2 * 16 + m] = acc2[r];
      o[3 * 16 + m] = acc3[r];
    }
  }
}

// ---------------- deterministic batch-norm ----------------

__global__ __launch_bounds__(256)
void bn_partial_kernel(const float* __restrict__ out,
                       float* __restrict__ part, int nOut) {
  __shared__ float s_sum[256], s_sq[256];
  const int c = threadIdx.x & 63;
  const int g = threadIdx.x >> 6;          // 4 row-groups per block
  float sum = 0.f, sq = 0.f;
  for (int row = blockIdx.x * 4 + g; row < nOut; row += gridDim.x * 4) {
    float v = out[(size_t)row * COUT + c];
    sum += v;
    sq  += v * v;
  }
  s_sum[threadIdx.x] = sum;
  s_sq[threadIdx.x]  = sq;
  __syncthreads();
  if (threadIdx.x < 64) {
    float ts = s_sum[c] + s_sum[64 + c] + s_sum[128 + c] + s_sum[192 + c];
    float tq = s_sq[c]  + s_sq[64 + c]  + s_sq[128 + c]  + s_sq[192 + c];
    part[blockIdx.x * 128 + c]      = ts;
    part[blockIdx.x * 128 + 64 + c] = tq;
  }
}

__global__ void bn_finalize_kernel(const float* __restrict__ part,
                                   const float* __restrict__ gamma,
                                   const float* __restrict__ beta,
                                   float* __restrict__ ss, int nOut) {
  int c = threadIdx.x;
  if (c >= 64) return;
  float sum = 0.f, sq = 0.f;
  for (int b = 0; b < RED_BLOCKS; ++b) {
    sum += part[b * 128 + c];
    sq  += part[b * 128 + 64 + c];
  }
  float inv_n = 1.0f / (float)nOut;
  float mean  = sum * inv_n;
  float var   = sq * inv_n - mean * mean;
  float sc    = gamma[c] * rsqrtf(var + BN_EPS);
  ss[c]       = sc;
  ss[64 + c]  = beta[c] - mean * sc;
}

__global__ void bn_apply_kernel(float* __restrict__ out,
                                const float* __restrict__ ss, int total) {
  int i = blockIdx.x * blockDim.x + threadIdx.x;
  for (; i < total; i += gridDim.x * blockDim.x) {
    int c = i & 63;
    float v = out[i] * ss[c] + ss[64 + c];
    out[i] = v > 0.f ? v : 0.f;
  }
}

// ---------------- launcher ----------------

extern "C" void kernel_launch(void* const* d_in, const int* in_sizes, int n_in,
                              void* d_out, int out_size, void* d_ws, size_t ws_size,
                              hipStream_t stream) {
  const float* features      = (const float*)d_in[0];
  const float* weight        = (const float*)d_in[1];
  const float* gamma         = (const float*)d_in[2];
  const float* beta          = (const float*)d_in[3];
  const int*   positions     = (const int*)d_in[4];
  const int*   out_positions = (const int*)d_in[5];
  float* out = (float*)d_out;

  const int nIn  = in_sizes[0] / CIN;
  const int nOut = in_sizes[5] / 3;

  // Carve workspace (256B-aligned slices): grid 8.79MB, feat16 ~6.2MB,
  // repacked weights 0.11MB, BN partials 0.13MB, BN scale/shift 512B.
  char* p = (char*)d_ws;
  auto carve = [&](size_t bytes) {
    char* r = p;
    p += (bytes + 255) & ~(size_t)255;
    return (void*)r;
  };
  int*      grid   = (int*)     carve((size_t)GDIM * GDIM * GDIM * sizeof(int));
  _Float16* feat16 = (_Float16*)carve((size_t)(nIn + 1) * CIN * sizeof(_Float16));
  _Float16* wB     = (_Float16*)carve((size_t)KVOL * 4 * 32 * 16 * sizeof(_Float16));
  float*    part   = (float*)   carve((size_t)RED_BLOCKS * 128 * sizeof(float));
  float*    ss     = (float*)   carve(128 * sizeof(float));

  const int gcells = GDIM * GDIM * GDIM;
  init_grid_kernel<<<512, 256, 0, stream>>>(grid, gcells);
  scatter_grid_kernel<<<(nIn + 255) / 256, 256, 0, stream>>>(grid, positions, nIn);
  cvt_features_kernel<<<512, 256, 0, stream>>>(features, feat16, nIn);
  repack_weights_kernel<<<(KVOL * 4 * 32 * 16 + 255) / 256, 256, 0, stream>>>(weight, wB);

  const int ntiles = (nOut + 15) / 16;
  const int blocks = (ntiles + CONV_WAVES - 1) / CONV_WAVES;
  const size_t ldsBytes = (size_t)KVOL * 4 * 32 * 16 * sizeof(_Float16); // 110592
  sparse_conv_wmma_kernel<<<blocks, CONV_WAVES * 32, ldsBytes, stream>>>(
      grid, feat16, wB, out_positions, out, nIn, nOut, ntiles);

  bn_partial_kernel<<<RED_BLOCKS, 256, 0, stream>>>(out, part, nOut);
  bn_finalize_kernel<<<1, 64, 0, stream>>>(part, gamma, beta, ss, nOut);
  bn_apply_kernel<<<512, 256, 0, stream>>>(out, ss, nOut * COUT);
}
